// MappingWithBias_42073499631906
// MI455X (gfx1250) — compile-verified, matrix operand order checked by reading, not run on hardware
//
#include <hip/hip_runtime.h>
#include <hip/hip_bf16.h>

// ---------------------------------------------------------------------------
// MappingWithBias for MI455X (gfx1250): WMMA f16 16x16x32 for the big MLP.
// B=128, N1=21, N2=4096, C=256, H=128, K(nn)=64, Cin=387 (pad 416).
// ---------------------------------------------------------------------------

typedef __attribute__((ext_vector_type(16))) _Float16 v16h;
typedef __attribute__((ext_vector_type(8)))  _Float16 v8h;
typedef __attribute__((ext_vector_type(8)))  float    v8f;

namespace cfg {
constexpr int Bsz  = 128;
constexpr int N1   = 21;
constexpr int N2   = 4096;
constexpr int C    = 256;   // points2 channels
constexpr int H    = 128;   // biasfold hidden
constexpr int KNN  = 64;
constexpr int CIN1 = 387;   // 128 + 256 + 3
constexpr int KP1  = 416;   // CIN1 padded to 32
constexpr int KT1  = KP1 / 32;  // 13
constexpr int KT2  = 256 / 32;  // 8
constexpr int NT   = 16;        // 256/16 output-channel tiles
constexpr int BN   = Bsz * N1;  // 2688 rows

// workspace layout (bytes, 256-aligned by construction)
constexpr size_t OFF_P1   = 0;                                   // BN*128 f32
constexpr size_t OFF_KNN  = OFF_P1  + (size_t)BN * H * 4;        // BN*64 i32
constexpr size_t OFF_WT1  = OFF_KNN + (size_t)BN * KNN * 4;      // 13*16*512 f16
constexpr size_t OFF_WT2  = OFF_WT1 + (size_t)KT1 * NT * 512 * 2;
constexpr size_t OFF_WT3  = OFF_WT2 + (size_t)KT2 * NT * 512 * 2;
constexpr size_t OFF_POOL = OFF_WT3 + (size_t)KT2 * NT * 512 * 2; // BN*256 f32
}

__device__ __forceinline__ float lrelu(float x) { return x > 0.0f ? x : 0.1f * x; }

__device__ __forceinline__ v16h cat16(v8h lo, v8h hi) {
  return __builtin_shufflevector(lo, hi, 0,1,2,3,4,5,6,7,8,9,10,11,12,13,14,15);
}

// ---------------------------------------------------------------------------
// Kernel 0: repack MLP weights into fragment-ready f16 tiles.
// Tile (kt, nt): 32 lanes x 16 halfs contiguous. For lane L (h = L>>4,
// ncol = nt*16 + (L&15)) element j maps to K = kt*32 + (j>>3)*16 + h*8 + (j&7).
// Source is row-major [out=256][in=Cin]; B(k,n) = W[n*Cin + k], 0 if k>=Cin.
// ---------------------------------------------------------------------------
__device__ __forceinline__ void fill_frag(_Float16* dst, const float* src, int Cin, int t) {
  int j    = t & 15;
  int lane = (t >> 4) & 31;
  int tile = t >> 9;
  int nt   = tile & 15;
  int kt   = tile >> 4;
  int hh   = lane >> 4;
  int ncol = nt * 16 + (lane & 15);
  int Kk   = kt * 32 + ((j >> 3) << 4) + hh * 8 + (j & 7);
  dst[t] = (_Float16)((Kk < Cin) ? src[(size_t)ncol * Cin + Kk] : 0.0f);
}

__global__ void prep_weights(const float* __restrict__ w1, const float* __restrict__ w2,
                             const float* __restrict__ w3,
                             _Float16* __restrict__ wt1, _Float16* __restrict__ wt2,
                             _Float16* __restrict__ wt3) {
  constexpr int T1 = cfg::KT1 * cfg::NT * 512;  // 106496
  constexpr int T2 = cfg::KT2 * cfg::NT * 512;  // 65536
  int g = blockIdx.x * 256 + threadIdx.x;
  if (g < T1)               fill_frag(wt1, w1, cfg::CIN1, g);
  else if (g < T1 + T2)     fill_frag(wt2, w2, 256, g - T1);
  else if (g < T1 + 2 * T2) fill_frag(wt3, w3, 256, g - T1 - T2);
}

// ---------------------------------------------------------------------------
// Kernel 1: biasfold — 3 x (conv1x1 + per-joint bias + BN(eval) + LeakyReLU)
// one block per (b, n1), 128 threads (one per output channel).
// ---------------------------------------------------------------------------
__global__ void __launch_bounds__(128) biasfold_kernel(
    const float* __restrict__ points1,
    const float* __restrict__ w1, const float* __restrict__ bb1,
    const float* __restrict__ g1, const float* __restrict__ be1,
    const float* __restrict__ w2, const float* __restrict__ bb2,
    const float* __restrict__ g2, const float* __restrict__ be2,
    const float* __restrict__ w3, const float* __restrict__ bb3,
    const float* __restrict__ g3, const float* __restrict__ be3,
    float* __restrict__ p1g) {
  __shared__ float xin[256];
  __shared__ float h1[128];
  __shared__ float h2[128];
  const int bn = blockIdx.x, b = bn / cfg::N1, n1 = bn % cfg::N1;
  const int o = threadIdx.x;
  const float bns = rsqrtf(1.0f + 1e-5f);

  for (int c = o; c < 256; c += 128)
    xin[c] = points1[((size_t)b * 256 + c) * cfg::N1 + n1];
  __syncthreads();

  float acc = bb1[o * cfg::N1 + n1];
  for (int c = 0; c < 256; ++c) acc += w1[o * 256 + c] * xin[c];
  h1[o] = lrelu(acc * (g1[o] * bns) + be1[o]);
  __syncthreads();

  acc = bb2[o * cfg::N1 + n1];
  for (int c = 0; c < 128; ++c) acc += w2[o * 128 + c] * h1[c];
  h2[o] = lrelu(acc * (g2[o] * bns) + be2[o]);
  __syncthreads();

  acc = bb3[o * cfg::N1 + n1];
  for (int c = 0; c < 128; ++c) acc += w3[o * 128 + c] * h2[c];
  p1g[(size_t)bn * 128 + o] = lrelu(acc * (g3[o] * bns) + be3[o]);
}

// ---------------------------------------------------------------------------
// Kernel 2: kNN (64 smallest squared distances out of 4096) per (b, n1).
// ---------------------------------------------------------------------------
__global__ void __launch_bounds__(256) knn_kernel(const float* __restrict__ xyz1,
                                                  const float* __restrict__ xyz2,
                                                  int* __restrict__ knng) {
  __shared__ float dist[cfg::N2];
  __shared__ float rv[256];
  __shared__ int   ri[256];
  const int bn = blockIdx.x, b = bn / cfg::N1, n1 = bn % cfg::N1;
  const int tid = threadIdx.x;

  const float px = xyz1[((size_t)b * 3 + 0) * cfg::N1 + n1];
  const float py = xyz1[((size_t)b * 3 + 1) * cfg::N1 + n1];
  const float pz = xyz1[((size_t)b * 3 + 2) * cfg::N1 + n1];
  const float* x2x = xyz2 + (size_t)b * 3 * cfg::N2;

  for (int j = tid; j < cfg::N2; j += 256) {
    float dx = x2x[j] - px;
    float dy = x2x[cfg::N2 + j] - py;
    float dz = x2x[2 * cfg::N2 + j] - pz;
    dist[j] = dx * dx + dy * dy + dz * dz;
  }
  __syncthreads();

  for (int k = 0; k < cfg::KNN; ++k) {
    float bv = 3.402823466e38f;
    int bi = 0;
    for (int j = tid; j < cfg::N2; j += 256)
      if (dist[j] < bv) { bv = dist[j]; bi = j; }
    rv[tid] = bv; ri[tid] = bi;
    __syncthreads();
    for (int s = 128; s > 0; s >>= 1) {
      if (tid < s && rv[tid + s] < rv[tid]) { rv[tid] = rv[tid + s]; ri[tid] = ri[tid + s]; }
      __syncthreads();
    }
    if (tid == 0) {
      knng[(size_t)bn * cfg::KNN + k] = ri[0];
      dist[ri[0]] = 3.402823466e38f;
    }
    __syncthreads();
  }
}

// ---------------------------------------------------------------------------
// Kernel 3: big MLP via WMMA + K-max-pool. One block (8 waves) per (b, n1).
// ---------------------------------------------------------------------------
__device__ __forceinline__ void mlp_layer(
    const _Float16* __restrict__ A, int kpad, int KT,
    const _Float16* __restrict__ W, const float* __restrict__ bias,
    _Float16* __restrict__ Out,
    int mt, int nh, int h, int nlo, int lane) {
  v8f acc[8];
  v8f z; 
#pragma unroll
  for (int i = 0; i < 8; ++i) z[i] = 0.0f;
#pragma unroll
  for (int t = 0; t < 8; ++t) acc[t] = z;

  const _Float16* arow = A + (size_t)(mt * 16 + nlo) * kpad;
  for (int kt = 0; kt < KT; ++kt) {
    const _Float16* pa = arow + kt * 32 + h * 8;
    v8h alo = *(const v8h*)pa;
    v8h ahi = *(const v8h*)(pa + 16);
    v16h av = cat16(alo, ahi);
    const _Float16* pb0 = W + ((size_t)(kt * cfg::NT + nh * 8) * 32 + lane) * 16;
#pragma unroll
    for (int t = 0; t < 8; ++t) {
      const _Float16* pb = pb0 + t * 512;
      v8h blo = *(const v8h*)pb;
      v8h bhi = *(const v8h*)(pb + 8);
      v16h bv = cat16(blo, bhi);
      acc[t] = __builtin_amdgcn_wmma_f32_16x16x32_f16(
          false, av, false, bv, (short)0, acc[t], false, false);
    }
  }
#pragma unroll
  for (int t = 0; t < 8; ++t) {
    int n = (nh * 8 + t) * 16 + nlo;
    float bvs = bias[n];
#pragma unroll
    for (int v = 0; v < 8; ++v) {
      int m = mt * 16 + v + 8 * h;  // C/D layout: VGPR v -> M=v (+8 for lanes 16-31)
      Out[m * 256 + n] = (_Float16)lrelu(acc[t][v] + bvs);
    }
  }
}

__global__ void __launch_bounds__(256) bigmlp_kernel(
    const float* __restrict__ xyz1, const float* __restrict__ xyz2,
    const float* __restrict__ points2, const float* __restrict__ p1g,
    const int* __restrict__ knng,
    const _Float16* __restrict__ wt1, const _Float16* __restrict__ wt2,
    const _Float16* __restrict__ wt3,
    const float* __restrict__ b1, const float* __restrict__ b2,
    const float* __restrict__ b3,
    float* __restrict__ pooled) {
  extern __shared__ char smem3[];
  _Float16* feat = (_Float16*)smem3;                 // [64][416]
  _Float16* actA = feat + 64 * cfg::KP1;             // [64][256]
  _Float16* actB = actA + 64 * 256;                  // [64][256]
  int* sidx = (int*)(actB + 64 * 256);               // [64]

  const int bn = blockIdx.x, b = bn / cfg::N1, n1 = bn % cfg::N1;
  const int tid = threadIdx.x;

  if (tid < 64) sidx[tid] = knng[(size_t)bn * cfg::KNN + tid];
  __syncthreads();

  // feat columns: [0,128) = p1 broadcast, [128,384) = gathered p2,
  //               [384,387) = direction xyz, [387,416) = zero pad
  const float* p1row = p1g + (size_t)bn * 128;
  for (int e = tid; e < 64 * 128; e += 256) {
    int r = e >> 7, c = e & 127;
    feat[r * cfg::KP1 + c] = (_Float16)p1row[c];
  }
  for (int e = tid; e < 64 * 256; e += 256) {
    int r = e >> 8, c = e & 255;
    feat[r * cfg::KP1 + 128 + c] =
        (_Float16)points2[((size_t)b * cfg::C + c) * cfg::N2 + sidx[r]];
  }
  for (int e = tid; e < 64 * 3; e += 256) {
    int r = e / 3, d = e % 3;
    float v = xyz2[((size_t)b * 3 + d) * cfg::N2 + sidx[r]] -
              xyz1[((size_t)b * 3 + d) * cfg::N1 + n1];
    feat[r * cfg::KP1 + 384 + d] = (_Float16)v;
  }
  for (int e = tid; e < 64 * (cfg::KP1 - cfg::CIN1); e += 256) {
    int r = e / 29, c = cfg::CIN1 + e % 29;
    feat[r * cfg::KP1 + c] = (_Float16)0.0f;
  }
  __syncthreads();

  const int lane = tid & 31, wid = tid >> 5;
  const int mt = wid & 3, nh = wid >> 2;
  const int h = lane >> 4, nlo = lane & 15;

  mlp_layer(feat, cfg::KP1, cfg::KT1, wt1, b1, actA, mt, nh, h, nlo, lane);
  __syncthreads();
  mlp_layer(actA, 256, cfg::KT2, wt2, b2, actB, mt, nh, h, nlo, lane);
  __syncthreads();
  mlp_layer(actB, 256, cfg::KT2, wt3, b3, actA, mt, nh, h, nlo, lane);
  __syncthreads();

  // max over the 64 neighbors (rows)
  float m = -3.402823466e38f;
  for (int r = 0; r < 64; ++r) m = fmaxf(m, (float)actA[r * 256 + tid]);
  pooled[(size_t)bn * 256 + tid] = m;
}

// ---------------------------------------------------------------------------
// Kernel 4: mlp2 (2 x 256x256, no bias, lrelu) + regress (3x256 + bias).
// ---------------------------------------------------------------------------
__global__ void __launch_bounds__(256) tail_kernel(
    const float* __restrict__ pooled,
    const float* __restrict__ w1, const float* __restrict__ w2,
    const float* __restrict__ rw, const float* __restrict__ rb,
    float* __restrict__ out) {
  __shared__ float a0[256];
  __shared__ float a1[256];
  __shared__ float a2[256];
  const int bn = blockIdx.x, b = bn / cfg::N1, n1 = bn % cfg::N1;
  const int o = threadIdx.x;

  a0[o] = pooled[(size_t)bn * 256 + o];
  __syncthreads();
  float acc = 0.0f;
  for (int c = 0; c < 256; ++c) acc += w1[o * 256 + c] * a0[c];
  a1[o] = lrelu(acc);
  __syncthreads();
  acc = 0.0f;
  for (int c = 0; c < 256; ++c) acc += w2[o * 256 + c] * a1[c];
  a2[o] = lrelu(acc);
  __syncthreads();
  if (o < 3) {
    float r = rb[o];
    for (int c = 0; c < 256; ++c) r += rw[o * 256 + c] * a2[c];
    out[((size_t)b * 3 + o) * cfg::N1 + n1] = r;  // [B,3,N1]
  }
}

// ---------------------------------------------------------------------------
extern "C" void kernel_launch(void* const* d_in, const int* in_sizes, int n_in,
                              void* d_out, int out_size, void* d_ws, size_t ws_size,
                              hipStream_t stream) {
  const float* xyz1    = (const float*)d_in[0];
  const float* xyz2    = (const float*)d_in[1];
  const float* points1 = (const float*)d_in[2];
  const float* points2 = (const float*)d_in[3];
  const float* bf_w1 = (const float*)d_in[4],  *bf_bb1 = (const float*)d_in[5];
  const float* bf_g1 = (const float*)d_in[6],  *bf_be1 = (const float*)d_in[7];
  const float* bf_w2 = (const float*)d_in[8],  *bf_bb2 = (const float*)d_in[9];
  const float* bf_g2 = (const float*)d_in[10], *bf_be2 = (const float*)d_in[11];
  const float* bf_w3 = (const float*)d_in[12], *bf_bb3 = (const float*)d_in[13];
  const float* bf_g3 = (const float*)d_in[14], *bf_be3 = (const float*)d_in[15];
  const float* mlp_w1 = (const float*)d_in[16], *mlp_b1 = (const float*)d_in[17];
  const float* mlp_w2 = (const float*)d_in[18], *mlp_b2 = (const float*)d_in[19];
  const float* mlp_w3 = (const float*)d_in[20], *mlp_b3 = (const float*)d_in[21];
  const float* mlp2_w1 = (const float*)d_in[22];
  const float* mlp2_w2 = (const float*)d_in[23];
  const float* reg_w = (const float*)d_in[24], *reg_b = (const float*)d_in[25];

  char* ws = (char*)d_ws;
  float*     p1g    = (float*)(ws + cfg::OFF_P1);
  int*       knng   = (int*)(ws + cfg::OFF_KNN);
  _Float16*  wt1    = (_Float16*)(ws + cfg::OFF_WT1);
  _Float16*  wt2    = (_Float16*)(ws + cfg::OFF_WT2);
  _Float16*  wt3    = (_Float16*)(ws + cfg::OFF_WT3);
  float*     pooled = (float*)(ws + cfg::OFF_POOL);
  float*     out    = (float*)d_out;

  constexpr int prepTotal = cfg::KT1 * cfg::NT * 512 + 2 * cfg::KT2 * cfg::NT * 512;
  prep_weights<<<(prepTotal + 255) / 256, 256, 0, stream>>>(mlp_w1, mlp_w2, mlp_w3,
                                                            wt1, wt2, wt3);
  biasfold_kernel<<<cfg::BN, 128, 0, stream>>>(points1,
      bf_w1, bf_bb1, bf_g1, bf_be1,
      bf_w2, bf_bb2, bf_g2, bf_be2,
      bf_w3, bf_bb3, bf_g3, bf_be3, p1g);
  knn_kernel<<<cfg::BN, 256, 0, stream>>>(xyz1, xyz2, knng);

  size_t smem = (size_t)(64 * cfg::KP1 + 2 * 64 * 256) * sizeof(_Float16) +
                64 * sizeof(int);
  bigmlp_kernel<<<cfg::BN, 256, smem, stream>>>(xyz1, xyz2, points2, p1g, knng,
                                                wt1, wt2, wt3,
                                                mlp_b1, mlp_b2, mlp_b3, pooled);
  tail_kernel<<<cfg::BN, 256, 0, stream>>>(pooled, mlp2_w1, mlp2_w2, reg_w, reg_b,
                                           out);
}
